// MoELayer_41918880809691
// MI455X (gfx1250) — matrix operand order repaired and checked by reading, not compiled, not run on hardware
//
#include <hip/hip_runtime.h>
#include <hip/hip_bf16.h>

// ---------- problem constants (from reference) ----------
#define T_TOK 8192      // B*L
#define Dm    1024
#define Em    8
#define Hm    512
#define MT    32        // tokens per MLP tile
#define TILES_PER_E (T_TOK / MT)   // 256

typedef __attribute__((ext_vector_type(4)))  unsigned int v4u;
typedef __attribute__((ext_vector_type(8)))  unsigned int v8u;
typedef __attribute__((ext_vector_type(16))) __bf16      v16bf;
typedef __attribute__((ext_vector_type(8)))  float       v8f;

static __device__ __forceinline__ float gelu_exact(float v) {
    return 0.5f * v * (1.0f + erff(v * 0.70710678118654752f));
}

// A operand: 16 contiguous K values per lane (two adjacent 16B runs)
static __device__ __forceinline__ v16bf load_a(const char* p) {
    v4u a0 = *(const v4u*)p;
    v4u a1 = *(const v4u*)(p + 16);
    v8u au = __builtin_shufflevector(a0, a1, 0, 1, 2, 3, 4, 5, 6, 7);
    return __builtin_bit_cast(v16bf, au);
}
// B operand: 16 contiguous K values per lane (32B)
static __device__ __forceinline__ v16bf load_b(const __bf16* p) {
    v8u bu = *(const v8u*)p;
    return __builtin_bit_cast(v16bf, bu);
}
static __device__ __forceinline__ v8f wmma_bf16(v16bf a, v16bf b, v8f c) {
    return __builtin_amdgcn_wmma_f32_16x16x32_bf16(
        false, a, false, b, (short)0, c, false, false);
}

// ---------------- kernel 1: zero output + expert counters ----------------
__global__ void k_init(float* __restrict__ out, int* __restrict__ cnt) {
    size_t i = (size_t)blockIdx.x * blockDim.x + threadIdx.x;
    size_t n = (size_t)T_TOK * Dm;
    size_t stride = (size_t)gridDim.x * blockDim.x;
    for (size_t j = i; j < n; j += stride) out[j] = 0.0f;
    if (i < Em) cnt[i] = 0;
}

// ---------------- kernel 2: convert x -> bf16 (row major) ----------------
__global__ void k_cvt_x(const float* __restrict__ x, __bf16* __restrict__ xb) {
    size_t i = (size_t)blockIdx.x * blockDim.x + threadIdx.x;
    size_t n = (size_t)T_TOK * Dm;
    size_t stride = (size_t)gridDim.x * blockDim.x;
    for (size_t j = i; j < n; j += stride) xb[j] = (__bf16)x[j];
}

// ------- kernel 3: convert+transpose weights: w[e][k][n] -> wt[e][n][k] ------
__global__ void k_cvt_w(const float* __restrict__ w, __bf16* __restrict__ wt,
                        int K, int N) {
    size_t i = (size_t)blockIdx.x * blockDim.x + threadIdx.x;
    size_t total = (size_t)Em * K * N;
    size_t stride = (size_t)gridDim.x * blockDim.x;
    for (size_t j = i; j < total; j += stride) {
        size_t kn = (size_t)K * N;
        size_t e = j / kn;
        size_t r = j - e * kn;
        size_t n = r / K;
        size_t k = r - n * K;
        wt[j] = (__bf16)w[e * kn + k * N + n];
    }
}

// ---------------- kernel 4: gating (one wave32 per token) ----------------
__global__ void k_gate(const float* __restrict__ x,
                       const float* __restrict__ gw,
                       const float* __restrict__ gb,
                       int* __restrict__ cnt,
                       int* __restrict__ tlist,
                       float* __restrict__ glist) {
    int wave = (blockIdx.x * blockDim.x + threadIdx.x) >> 5;
    int lane = threadIdx.x & 31;
    if (wave >= T_TOK) return;

    const float* xr = x + (size_t)wave * Dm;
    float acc[Em];
#pragma unroll
    for (int e = 0; e < Em; ++e) acc[e] = 0.0f;

    for (int d = lane; d < Dm; d += 32) {
        float xv = xr[d];
        const float* g = gw + (size_t)d * Em;
#pragma unroll
        for (int e = 0; e < Em; ++e) acc[e] += xv * g[e];
    }
#pragma unroll
    for (int off = 16; off > 0; off >>= 1) {
#pragma unroll
        for (int e = 0; e < Em; ++e)
            acc[e] += __shfl_xor(acc[e], off, 32);
    }
    if (lane == 0) {
#pragma unroll
        for (int e = 0; e < Em; ++e) acc[e] += gb[e];
        int i0 = 0;
#pragma unroll
        for (int e = 1; e < Em; ++e) if (acc[e] > acc[i0]) i0 = e;
        int i1 = (i0 == 0) ? 1 : 0;
#pragma unroll
        for (int e = 0; e < Em; ++e)
            if (e != i0 && acc[e] > acc[i1]) i1 = e;
        float v0 = acc[i0], v1 = acc[i1];
        float e1 = __expf(v1 - v0);        // v0 is max
        float s  = 1.0f / (1.0f + e1);
        float g0 = s, g1 = e1 * s;
        int p0 = atomicAdd(&cnt[i0], 1);
        tlist[i0 * T_TOK + p0] = wave;
        glist[i0 * T_TOK + p0] = g0;
        int p1 = atomicAdd(&cnt[i1], 1);
        tlist[i1 * T_TOK + p1] = wave;
        glist[i1 * T_TOK + p1] = g1;
    }
}

// ---------------- kernel 5: grouped expert MLP (WMMA bf16) ----------------
// grid: Em * TILES_PER_E blocks, 256 threads (8 wave32).
// Each wave owns n-tiles {wv, wv+8, wv+16, wv+24} and BOTH 16-row groups,
// so every B tile feeds 2 WMMAs (halves weight traffic vs. 1:1).
__global__ __launch_bounds__(256) void k_mlp(
        const __bf16* __restrict__ xb,    // [T][D] bf16
        const __bf16* __restrict__ w1t,   // [E][H][D] bf16 (pre-transposed)
        const __bf16* __restrict__ w2t,   // [E][D][H] bf16 (pre-transposed)
        const float*  __restrict__ b1,    // [E][H]
        const float*  __restrict__ b2,    // [E][D]
        const int*    __restrict__ cnt,
        const int*    __restrict__ tlist, // [E][T]
        const float*  __restrict__ glist, // [E][T]
        float* __restrict__ out) {        // [T][D]

    __shared__ __bf16 xs[MT * 272];   // 32 rows x (256 K-chunk + 16 pad) = 17,408 B
    __shared__ __bf16 hs[MT * 528];   // 32 rows x (512 + 16 pad)         = 33,792 B
    __shared__ int    tok[MT];
    __shared__ float  gts[MT];

    const int e    = blockIdx.x >> 8;          // / TILES_PER_E
    const int tile = blockIdx.x & (TILES_PER_E - 1);
    const int count = cnt[e];
    if (tile * MT >= count) return;            // uniform exit

    const int tid  = threadIdx.x;
    const int wv   = tid >> 5;
    const int lane = tid & 31;
    const int lm   = lane & 15;                // N (B/C) and M (A) index
    const int lh   = lane >> 4;                // K-half selector

    if (tid < MT) {
        int r = tile * MT + tid;
        bool valid = r < count;
        tok[tid] = tlist[e * T_TOK + (valid ? r : 0)];
        gts[tid] = valid ? glist[e * T_TOK + r] : 0.0f;
    }
    __syncthreads();

    const __bf16* w1e = w1t + (size_t)e * Hm * Dm;
    const __bf16* w2e = w2t + (size_t)e * Dm * Hm;
    const v8f zero8 = {0.f,0.f,0.f,0.f,0.f,0.f,0.f,0.f};

    // ---------------- layer 1: h = gelu(x @ w1 + b1) ----------------
    v8f acc[4][2];
#pragma unroll
    for (int j = 0; j < 4; ++j) { acc[j][0] = zero8; acc[j][1] = zero8; }

    const char* a0base = (const char*)xs + (size_t)lm * 544;         // rows 0..15
    const char* a1base = (const char*)xs + (size_t)(16 + lm) * 544;  // rows 16..31

    for (int kc = 0; kc < Dm / 256; ++kc) {
        __syncthreads();   // protect xs before overwrite
        {   // cooperative stage of x[:, kc*256 .. +256) for the 32 tile rows
            int row = tid >> 3, seg = tid & 7;
            const v4u* src = (const v4u*)(xb + (size_t)tok[row] * Dm + kc * 256) + seg * 4;
            v4u* dst = (v4u*)((char*)xs + (size_t)row * 544) + seg * 4;
            dst[0] = src[0]; dst[1] = src[1]; dst[2] = src[2]; dst[3] = src[3];
        }
        __syncthreads();
#pragma unroll
        for (int ks = 0; ks < 8; ++ks) {
            v16bf A0 = load_a(a0base + ks * 64 + lh * 32);
            v16bf A1 = load_a(a1base + ks * 64 + lh * 32);
            int kglob = kc * 256 + ks * 32 + lh * 16;
#pragma unroll
            for (int j = 0; j < 4; ++j) {
                int n0 = (wv + 8 * j) * 16;
                v16bf Bm = load_b(w1e + (size_t)(n0 + lm) * Dm + kglob);
                acc[j][0] = wmma_bf16(A0, Bm, acc[j][0]);
                acc[j][1] = wmma_bf16(A1, Bm, acc[j][1]);
            }
        }
    }
    // epilogue: bias + gelu -> hs (bf16), both row groups
#pragma unroll
    for (int j = 0; j < 4; ++j) {
        int n0 = (wv + 8 * j) * 16;
        float bias = b1[e * Hm + n0 + lm];
#pragma unroll
        for (int mg = 0; mg < 2; ++mg) {
            int rowbase = mg * 16 + lh * 8;
#pragma unroll
            for (int r = 0; r < 8; ++r) {
                float v = acc[j][mg][r] + bias;
                hs[(size_t)(rowbase + r) * 528 + n0 + lm] = (__bf16)gelu_exact(v);
            }
        }
    }
    __syncthreads();

    // ---------------- layer 2: out += gate * (h @ w2 + b2) ----------------
    float grow[2][8];
    int   trow[2][8];
#pragma unroll
    for (int mg = 0; mg < 2; ++mg) {
        int rowbase = mg * 16 + lh * 8;
#pragma unroll
        for (int r = 0; r < 8; ++r) {
            grow[mg][r] = gts[rowbase + r];
            trow[mg][r] = tok[rowbase + r];
        }
    }
    const char* h0base = (const char*)hs + (size_t)lm * 1056;         // rows 0..15
    const char* h1base = (const char*)hs + (size_t)(16 + lm) * 1056;  // rows 16..31

    for (int pass = 0; pass < 2; ++pass) {
        v8f acc2[4][2];
#pragma unroll
        for (int j = 0; j < 4; ++j) { acc2[j][0] = zero8; acc2[j][1] = zero8; }
#pragma unroll
        for (int ks = 0; ks < Hm / 32; ++ks) {
            v16bf A0 = load_a(h0base + ks * 64 + lh * 32);
            v16bf A1 = load_a(h1base + ks * 64 + lh * 32);
#pragma unroll
            for (int j = 0; j < 4; ++j) {
                int n0 = (wv + 8 * (4 * pass + j)) * 16;
                v16bf Bm = load_b(w2e + (size_t)(n0 + lm) * Hm + ks * 32 + lh * 16);
                acc2[j][0] = wmma_bf16(A0, Bm, acc2[j][0]);
                acc2[j][1] = wmma_bf16(A1, Bm, acc2[j][1]);
            }
        }
#pragma unroll
        for (int j = 0; j < 4; ++j) {
            int n0 = (wv + 8 * (4 * pass + j)) * 16;
            float bias = b2[e * Dm + n0 + lm];
#pragma unroll
            for (int mg = 0; mg < 2; ++mg) {
#pragma unroll
                for (int r = 0; r < 8; ++r) {
                    float v = (acc2[j][mg][r] + bias) * grow[mg][r];
                    unsafeAtomicAdd(out + (size_t)trow[mg][r] * Dm + n0 + lm, v);
                }
            }
        }
    }
}

// ---------------------------- host launcher ----------------------------
extern "C" void kernel_launch(void* const* d_in, const int* in_sizes, int n_in,
                              void* d_out, int out_size, void* d_ws, size_t ws_size,
                              hipStream_t stream) {
    const float* x  = (const float*)d_in[0];
    const float* gw = (const float*)d_in[1];
    const float* gb = (const float*)d_in[2];
    const float* w1 = (const float*)d_in[3];
    const float* b1 = (const float*)d_in[4];
    const float* w2 = (const float*)d_in[5];
    const float* b2 = (const float*)d_in[6];
    float* out = (float*)d_out;

    char* ws = (char*)d_ws;
    __bf16* xb  = (__bf16*)ws;  ws += (size_t)T_TOK * Dm * 2;
    __bf16* w1t = (__bf16*)ws;  ws += (size_t)Em * Dm * Hm * 2;
    __bf16* w2t = (__bf16*)ws;  ws += (size_t)Em * Dm * Hm * 2;
    int*    tlist = (int*)ws;   ws += (size_t)Em * T_TOK * 4;
    float*  glist = (float*)ws; ws += (size_t)Em * T_TOK * 4;
    int*    cnt   = (int*)ws;   ws += 256;

    k_init <<<2048, 256, 0, stream>>>(out, cnt);
    k_cvt_x<<<2048, 256, 0, stream>>>(x, xb);
    k_cvt_w<<<2048, 256, 0, stream>>>(w1, w1t, Dm, Hm);   // -> [e][n<H][k<D]
    k_cvt_w<<<2048, 256, 0, stream>>>(w2, w2t, Hm, Dm);   // -> [e][n<D][k<H]
    k_gate <<<T_TOK / 8, 256, 0, stream>>>(x, gw, gb, cnt, tlist, glist);
    k_mlp  <<<Em * TILES_PER_E, 256, 0, stream>>>(xb, w1t, w2t, b1, b2,
                                                  cnt, tlist, glist, out);
}
